// DotProductAttention_22789096472738
// MI455X (gfx1250) — matrix operand order repaired and checked by reading, not compiled, not run on hardware
//
#include <hip/hip_runtime.h>
#include <hip/hip_bf16.h>

typedef __bf16 bf16_t;
typedef __attribute__((ext_vector_type(16))) __bf16 v16bf;
typedef __attribute__((ext_vector_type(8)))  __bf16 v8bf;
typedef __attribute__((ext_vector_type(8)))  float  v8f;

#define B_ 2
#define S_ 4096
#define D_ 512

// ---------------------------------------------------------------------------
// WMMA fragment helpers (wave32); A/B fragments hold k = 16*(lane/16) + s (contiguous order).
// ---------------------------------------------------------------------------

static __device__ __forceinline__ v16bf cat8(v8bf lo, v8bf hi) {
  return __builtin_shufflevector(lo, hi, 0,1,2,3,4,5,6,7,8,9,10,11,12,13,14,15);
}

// A-fragment 16x32 bf16 from a row-major bf16 matrix.
static __device__ __forceinline__ v16bf load_a_bf16(const bf16_t* base, int pitch,
                                                    int row0, int k0, int lane) {
  const int row = row0 + (lane & 15);
  const int off = (lane >> 4) << 4;                 // 0 or 16
  const bf16_t* p = base + (size_t)row * pitch + k0 + off;
  v8bf lo = *(const v8bf*)p;                        // 16B aligned
  v8bf hi = *(const v8bf*)(p + 8);
  return cat8(lo, hi);
}

// Same A-fragment but sourced from fp32 (converted in-register to bf16).
static __device__ __forceinline__ v16bf load_a_f32(const float* base, int pitch,
                                                   int row0, int k0, int lane) {
  const int row = row0 + (lane & 15);
  const int off = (lane >> 4) << 4;
  const float* p0 = base + (size_t)row * pitch + k0 + off;
  const float* p1 = p0 + 8;
  v16bf a;
#pragma unroll
  for (int j = 0; j < 8; ++j) { a[j] = (__bf16)p0[j]; a[j + 8] = (__bf16)p1[j]; }
  return a;
}

// B-fragment 32x16 bf16 from "column-major" storage colmaj[n][k] (contig K).
static __device__ __forceinline__ v16bf load_b_bf16(const bf16_t* colmaj, int pitch,
                                                    int n0, int k0, int lane) {
  const int n    = n0 + (lane & 15);
  const int koff = (lane >> 4) << 4;                // 0 or 16
  const bf16_t* p = colmaj + (size_t)n * pitch + k0 + koff;
  v8bf lo = *(const v8bf*)p;
  v8bf hi = *(const v8bf*)(p + 8);
  return cat8(lo, hi);
}

// B-fragment from an LDS tile laid out [64 cols][32 k] bf16.
static __device__ __forceinline__ v16bf load_b_lds(const bf16_t* tile, int n0, int lane) {
  const int n    = n0 + (lane & 15);
  const int koff = (lane >> 4) << 4;
  const bf16_t* p = tile + n * 32 + koff;
  v8bf lo = *(const v8bf*)p;                        // ds_load_b128
  v8bf hi = *(const v8bf*)(p + 8);
  return cat8(lo, hi);
}

// C/D 16x16 f32: lane<16 -> N=lane, M=r; lane>=16 -> N=lane-16, M=r+8.
static __device__ __forceinline__ void store_c_f32(float* base, int pitch, int row0,
                                                   int col0, int lane, v8f c) {
  const int col = col0 + (lane & 15);
  const int r0  = row0 + ((lane >> 4) << 3);
#pragma unroll
  for (int r = 0; r < 8; ++r) base[(size_t)(r0 + r) * pitch + col] = c[r];
}

static __device__ __forceinline__ void store_c_bf16(bf16_t* base, int pitch, int row0,
                                                    int col0, int lane, v8f c) {
  const int col = col0 + (lane & 15);
  const int r0  = row0 + ((lane >> 4) << 3);
#pragma unroll
  for (int r = 0; r < 8; ++r) base[(size_t)(r0 + r) * pitch + col] = (__bf16)c[r];
}

static __device__ __forceinline__ v8f wmma_bf16(v16bf a, v16bf b, v8f c) {
  return __builtin_amdgcn_wmma_f32_16x16x32_bf16(false, a, false, b, (short)0, c,
                                                 false, false);
}

// ---------------------------------------------------------------------------
// Prep kernels: transpose-cast so every WMMA B-fragment load is contiguous.
// ---------------------------------------------------------------------------

// WT[n][k] = bf16(W[k][n]), 512x512
__global__ void k_transpose_w(const float* __restrict__ W, bf16_t* __restrict__ WT) {
  int idx = blockIdx.x * blockDim.x + threadIdx.x;
  if (idx >= D_ * D_) return;
  int n = idx >> 9, k = idx & (D_ - 1);
  WT[idx] = (__bf16)W[k * D_ + n];
}

// vT[b][d][s] = bf16(key[b][s][d])
__global__ void k_transpose_v(const float* __restrict__ key, bf16_t* __restrict__ vT) {
  int idx = blockIdx.x * blockDim.x + threadIdx.x;
  if (idx >= B_ * D_ * S_) return;
  int s = idx & (S_ - 1);
  int d = (idx >> 12) & (D_ - 1);
  int b = idx >> 21;
  vT[idx] = (__bf16)key[((size_t)b * S_ + s) * D_ + d];
}

// ---------------------------------------------------------------------------
// Projection GEMM: C[8192,512](bf16) = A[8192,512](f32) x W (via WT[n][k]).
// Block = 128 threads (4 waves); workgroup tile 64x64; wave tile 16x64.
// ---------------------------------------------------------------------------
__global__ void k_proj(const float* __restrict__ A, const bf16_t* __restrict__ WT,
                       bf16_t* __restrict__ C) {
  const int lane = threadIdx.x & 31;
  const int wave = threadIdx.x >> 5;
  const int row0 = blockIdx.y * 64 + wave * 16;
  const int col0 = blockIdx.x * 64;
  v8f acc0 = {}, acc1 = {}, acc2 = {}, acc3 = {};
  for (int k0 = 0; k0 < D_; k0 += 32) {
    v16bf a  = load_a_f32(A, D_, row0, k0, lane);
    v16bf b0 = load_b_bf16(WT, D_, col0,      k0, lane);
    v16bf b1 = load_b_bf16(WT, D_, col0 + 16, k0, lane);
    v16bf b2 = load_b_bf16(WT, D_, col0 + 32, k0, lane);
    v16bf b3 = load_b_bf16(WT, D_, col0 + 48, k0, lane);
    acc0 = wmma_bf16(a, b0, acc0);
    acc1 = wmma_bf16(a, b1, acc1);
    acc2 = wmma_bf16(a, b2, acc2);
    acc3 = wmma_bf16(a, b3, acc3);
  }
  store_c_bf16(C, D_, row0, col0,      lane, acc0);
  store_c_bf16(C, D_, row0, col0 + 16, lane, acc1);
  store_c_bf16(C, D_, row0, col0 + 32, lane, acc2);
  store_c_bf16(C, D_, row0, col0 + 48, lane, acc3);
}

// ---------------------------------------------------------------------------
// Scores GEMM: S[b][q][kk](f32) = sum_d qb[b][q][d] * kb[b][kk][d].
// Wave tile 32x64 (8 accumulators) -> each B fragment reused by two WMMAs.
// ---------------------------------------------------------------------------
__global__ void k_scores(const bf16_t* __restrict__ qb, const bf16_t* __restrict__ kb,
                         float* __restrict__ sc) {
  const int b = blockIdx.z;
  const bf16_t* Q = qb + (size_t)b * S_ * D_;
  const bf16_t* K = kb + (size_t)b * S_ * D_;
  float* Sb = sc + (size_t)b * S_ * S_;
  const int lane = threadIdx.x & 31;
  const int wave = threadIdx.x >> 5;
  const int row0 = blockIdx.y * 128 + wave * 32;
  const int col0 = blockIdx.x * 64;
  v8f acc[8] = {};
  for (int k0 = 0; k0 < D_; k0 += 32) {
    if (k0 + 32 < D_)
      __builtin_prefetch(Q + (size_t)(row0 + (lane & 15)) * D_ + k0 + 32, 0, 0);
    v16bf a0 = load_a_bf16(Q, D_, row0,      k0, lane);
    v16bf a1 = load_a_bf16(Q, D_, row0 + 16, k0, lane);
#pragma unroll
    for (int t = 0; t < 4; ++t) {
      v16bf bf = load_b_bf16(K, D_, col0 + 16 * t, k0, lane);
      acc[t]     = wmma_bf16(a0, bf, acc[t]);
      acc[t + 4] = wmma_bf16(a1, bf, acc[t + 4]);
    }
  }
#pragma unroll
  for (int t = 0; t < 4; ++t) {
    store_c_f32(Sb, S_, row0,      col0 + 16 * t, lane, acc[t]);
    store_c_f32(Sb, S_, row0 + 16, col0 + 16 * t, lane, acc[t + 4]);
  }
}

// ---------------------------------------------------------------------------
// Row softmax (unscaled, matching reference), fp32 in -> bf16 out IN PLACE.
// bf16 row r lives in the first 8KB of the 16KB fp32 row (elem pitch 8192).
// ---------------------------------------------------------------------------
__global__ void k_softmax(float* __restrict__ sc) {
  const int row = blockIdx.x;
  const int b   = blockIdx.y;
  float*  rowp = sc + (((size_t)b * S_) + row) * S_;
  bf16_t* outp = (bf16_t*)rowp;
  const int t = threadIdx.x;

  float v[16];
#pragma unroll
  for (int j = 0; j < 16; ++j) v[j] = rowp[t * 16 + j];

  float m = v[0];
#pragma unroll
  for (int j = 1; j < 16; ++j) m = fmaxf(m, v[j]);
#pragma unroll
  for (int o = 16; o > 0; o >>= 1) m = fmaxf(m, __shfl_xor(m, o, 32));

  __shared__ float smax[8], ssum[8];
  if ((t & 31) == 0) smax[t >> 5] = m;
  __syncthreads();
  float M = smax[0];
#pragma unroll
  for (int i = 1; i < 8; ++i) M = fmaxf(M, smax[i]);

  float s = 0.0f;
#pragma unroll
  for (int j = 0; j < 16; ++j) { v[j] = __expf(v[j] - M); s += v[j]; }
#pragma unroll
  for (int o = 16; o > 0; o >>= 1) s += __shfl_xor(s, o, 32);
  if ((t & 31) == 0) ssum[t >> 5] = s;
  __syncthreads();                 // also orders: all fp32 reads done before bf16 writes
  float Sum = ssum[0];
#pragma unroll
  for (int i = 1; i < 8; ++i) Sum += ssum[i];
  const float inv = 1.0f / Sum;

#pragma unroll
  for (int j = 0; j < 16; ++j) outp[t * 16 + j] = (__bf16)(v[j] * inv);
}

// ---------------------------------------------------------------------------
// Output GEMM: O[b][q][d](f32) = sum_k P[b][q][k] * key[b][k][d]  (V = key).
// Workgroup tile 128x64, wave tile 32x64. The shared 64x32 V-tile is staged
// in LDS via GLOBAL_LOAD_ASYNC_TO_LDS_B128 (ASYNCcnt), double-buffered.
// ---------------------------------------------------------------------------
__global__ void k_out(const bf16_t* __restrict__ P, const bf16_t* __restrict__ vT,
                      float* __restrict__ O) {
  __shared__ __align__(16) bf16_t tileB[2][64 * 32];   // 2 x 4 KB
  const int b = blockIdx.z;
  const bf16_t* Pb = P  + (size_t)b * S_ * (2 * S_);   // bf16 elem pitch 8192
  const bf16_t* Vb = vT + (size_t)b * D_ * S_;
  float* Ob = O + (size_t)b * S_ * D_;
  const int lane = threadIdx.x & 31;
  const int wave = threadIdx.x >> 5;
  const int row0 = blockIdx.y * 128 + wave * 32;
  const int col0 = blockIdx.x * 64;

  // Cooperative async fill: 256 chunks of 16B (4KB tile), 2 chunks per thread.
  const int c0 = threadIdx.x;
  const int c1 = threadIdx.x + 128;
  auto fill = [&](int buf, int k0) {
    const int colA = c0 >> 2, segA = c0 & 3;
    const int colB = c1 >> 2, segB = c1 & 3;
    const bf16_t* gA = Vb + (size_t)(col0 + colA) * S_ + k0 + segA * 8;
    const bf16_t* gB = Vb + (size_t)(col0 + colB) * S_ + k0 + segB * 8;
    unsigned lA = (unsigned)(uintptr_t)&tileB[buf][c0 * 8];   // LDS byte offset
    unsigned lB = (unsigned)(uintptr_t)&tileB[buf][c1 * 8];
    asm volatile("global_load_async_to_lds_b128 %0, %1, off"
                 :: "v"(lA), "v"(gA) : "memory");
    asm volatile("global_load_async_to_lds_b128 %0, %1, off"
                 :: "v"(lB), "v"(gB) : "memory");
  };

  v8f acc[8] = {};
  fill(0, 0);
  for (int step = 0; step < S_ / 32; ++step) {
    const int k0  = step * 32;
    const int cur = step & 1;
    asm volatile("s_wait_asynccnt 0x0" ::: "memory");  // my fills landed in LDS
    __syncthreads();                                   // everyone's fills visible
    if (step + 1 < S_ / 32) fill(cur ^ 1, k0 + 32);    // prefetch next tile

    if (k0 + 32 < S_)
      __builtin_prefetch(Pb + (size_t)(row0 + (lane & 15)) * (2 * S_) + k0 + 32, 0, 0);
    v16bf a0 = load_a_bf16(Pb, 2 * S_, row0,      k0, lane);
    v16bf a1 = load_a_bf16(Pb, 2 * S_, row0 + 16, k0, lane);
#pragma unroll
    for (int t = 0; t < 4; ++t) {
      v16bf bf = load_b_lds(&tileB[cur][0], 16 * t, lane);
      acc[t]     = wmma_bf16(a0, bf, acc[t]);
      acc[t + 4] = wmma_bf16(a1, bf, acc[t + 4]);
    }
  }
#pragma unroll
  for (int t = 0; t < 4; ++t) {
    store_c_f32(Ob, D_, row0,      col0 + 16 * t, lane, acc[t]);
    store_c_f32(Ob, D_, row0 + 16, col0 + 16 * t, lane, acc[t + 4]);
  }
}

// ---------------------------------------------------------------------------
extern "C" void kernel_launch(void* const* d_in, const int* in_sizes, int n_in,
                              void* d_out, int out_size, void* d_ws, size_t ws_size,
                              hipStream_t stream) {
  const float* query = (const float*)d_in[0];
  const float* key   = (const float*)d_in[1];
  const float* Wq    = (const float*)d_in[2];
  const float* Wk    = (const float*)d_in[3];
  float* out = (float*)d_out;

  char* ws = (char*)d_ws;
  const size_t qk_bytes = (size_t)B_ * S_ * D_ * sizeof(bf16_t);   // 8 MiB each
  const size_t w_bytes  = (size_t)D_ * D_ * sizeof(bf16_t);        // 0.5 MiB each
  bf16_t* qb  = (bf16_t*)(ws);
  bf16_t* kb  = (bf16_t*)(ws + qk_bytes);
  bf16_t* vbT = (bf16_t*)(ws + 2 * qk_bytes);
  bf16_t* WqT = (bf16_t*)(ws + 3 * qk_bytes);
  bf16_t* WkT = (bf16_t*)(ws + 3 * qk_bytes + w_bytes);
  float*  sc  = (float*)(ws + 3 * qk_bytes + 2 * w_bytes);         // 128 MiB

  // 1. transpose-cast weights and V(=key)
  k_transpose_w<<<(D_ * D_ + 255) / 256, 256, 0, stream>>>(Wq, WqT);
  k_transpose_w<<<(D_ * D_ + 255) / 256, 256, 0, stream>>>(Wk, WkT);
  k_transpose_v<<<(B_ * D_ * S_ + 255) / 256, 256, 0, stream>>>(key, vbT);

  // 2. projections (WMMA bf16, fp32 accumulate)
  dim3 gproj(D_ / 64, (B_ * S_) / 64);
  k_proj<<<gproj, 128, 0, stream>>>(query, WqT, qb);
  k_proj<<<gproj, 128, 0, stream>>>(key,   WkT, kb);

  // 3. scores = q k^T (fp32 out, L2-resident: 128 MiB < 192 MiB L2)
  dim3 gsc(S_ / 64, S_ / 128, B_);
  k_scores<<<gsc, 128, 0, stream>>>(qb, kb, sc);

  // 4. row softmax, bf16 probs written in place
  dim3 gsm(S_, B_);
  k_softmax<<<gsm, 256, 0, stream>>>(sc);

  // 5. out = probs @ key (async-LDS staged V tiles)
  dim3 gout(D_ / 64, S_ / 128, B_);
  k_out<<<gout, 128, 0, stream>>>((const bf16_t*)sc, vbT, out);
}
